// UnquantizedSparseMoELayer_8641474200288
// MI455X (gfx1250) — matrix operand order repaired and checked by reading, not compiled, not run on hardware
//
#include <hip/hip_runtime.h>

#define TT 4096
#define DD 1024
#define FF 2048
#define EE 8

typedef __attribute__((ext_vector_type(16))) __bf16 v16bf;
typedef __attribute__((ext_vector_type(8)))  __bf16 bf16x8;
typedef __attribute__((ext_vector_type(8)))  float  v8f;

// Routing lists + one-shot bf16 copies of activations/weights (re-built every
// launch; no reliance on ws_size). bf16 weights (~100MB) nearly fit in L2.
__device__ int    g_count[EE];
__device__ int    g_tok[EE][TT];
__device__ float  g_wt[EE][TT];
__device__ __bf16 gw1[(size_t)EE * 2 * FF * DD];   // 67 MB
__device__ __bf16 gw2[(size_t)EE * DD * FF];       // 33.5 MB
__device__ __bf16 gx[(size_t)TT * DD];             // 8.4 MB

// ---------------------------------------------------------------------------
// CDNA5 async copy: global -> LDS, 16B per lane, tracked by ASYNCcnt.
// ---------------------------------------------------------------------------
__device__ __forceinline__ void async_load16(unsigned lds_off, const __bf16* gptr) {
  asm volatile("global_load_async_to_lds_b128 %0, %1, off"
               :: "v"(lds_off), "v"(gptr) : "memory");
}
__device__ __forceinline__ void wait_async0() {
  asm volatile("s_wait_asynccnt 0" ::: "memory");
}

// ---------------------------------------------------------------------------
// Kernel 0: zero output + expert counters
// ---------------------------------------------------------------------------
__global__ void zero_kernel(float* __restrict__ out, int n) {
  int i = blockIdx.x * blockDim.x + threadIdx.x;
  if (i < n) out[i] = 0.0f;
  if (blockIdx.x == 0 && threadIdx.x < EE) g_count[threadIdx.x] = 0;
}

// ---------------------------------------------------------------------------
// Kernel 1: fp32 -> bf16 copies of w1, w2, x (8 elements / thread)
// ---------------------------------------------------------------------------
__device__ __forceinline__ void cvt8(__bf16* dst, const float* src) {
  float4 f0 = *(const float4*)(src);
  float4 f1 = *(const float4*)(src + 4);
  bf16x8 a;
  a[0] = (__bf16)f0.x; a[1] = (__bf16)f0.y; a[2] = (__bf16)f0.z; a[3] = (__bf16)f0.w;
  a[4] = (__bf16)f1.x; a[5] = (__bf16)f1.y; a[6] = (__bf16)f1.z; a[7] = (__bf16)f1.w;
  *(bf16x8*)(dst) = a;
}

__global__ void cvt_kernel(const float* __restrict__ x,
                           const float* __restrict__ w1,
                           const float* __restrict__ w2) {
  const size_t N1 = (size_t)EE * 2 * FF * DD;
  const size_t N2 = (size_t)EE * DD * FF;
  const size_t NX = (size_t)TT * DD;
  size_t i = ((size_t)blockIdx.x * blockDim.x + threadIdx.x) * 8;
  if (i < N1)           { cvt8(gw1 + i,             w1 + i);              return; }
  if (i < N1 + N2)      { cvt8(gw2 + (i - N1),      w2 + (i - N1));       return; }
  if (i < N1 + N2 + NX) { cvt8(gx  + (i - N1 - N2), x  + (i - N1 - N2));  return; }
}

// ---------------------------------------------------------------------------
// Kernel 2: softmax + top-2 + renormalize, compact into per-expert lists
// ---------------------------------------------------------------------------
__global__ void route_kernel(const float* __restrict__ gating) {
  int t = blockIdx.x * blockDim.x + threadIdx.x;
  if (t >= TT) return;
  float v[EE];
  float mx = -3.0e38f;
#pragma unroll
  for (int i = 0; i < EE; ++i) { v[i] = gating[t * EE + i]; mx = fmaxf(mx, v[i]); }
#pragma unroll
  for (int i = 0; i < EE; ++i) v[i] = __expf(v[i] - mx);
  int i1 = 0; float b1 = v[0];
#pragma unroll
  for (int i = 1; i < EE; ++i) if (v[i] > b1) { b1 = v[i]; i1 = i; }
  int i2 = -1; float b2 = -1.0f;
#pragma unroll
  for (int i = 0; i < EE; ++i) if (i != i1 && v[i] > b2) { b2 = v[i]; i2 = i; }
  float inv = __builtin_amdgcn_rcpf(b1 + b2);
  int p1 = atomicAdd(&g_count[i1], 1);
  g_tok[i1][p1] = t; g_wt[i1][p1] = b1 * inv;
  int p2 = atomicAdd(&g_count[i2], 1);
  g_tok[i2][p2] = t; g_wt[i2][p2] = b2 * inv;
}

// Assemble a 16-element bf16 fragment from two 16B LDS loads.
__device__ __forceinline__ v16bf frag2(const __bf16* lo, const __bf16* hi) {
  bf16x8 a = *(const bf16x8*)lo;
  bf16x8 b = *(const bf16x8*)hi;
  v16bf r;
#pragma unroll
  for (int i = 0; i < 8; ++i) { r[i] = a[i]; r[i + 8] = b[i]; }
  return r;
}

// LDS layout (byte offsets); all 16B aligned. Double-buffered weight staging.
#define SD   (DD + 8)            // 1032: xs row stride (bf16 elems)
#define SF   (FF + 8)            // 2056: hs row stride
#define SW2  72                  // weight tile row stride ([n][k], 64 k + pad)
#define XS_OFF   0u
#define HS_OFF   (XS_OFF  + 2u * 16 * SD)      //  33,024
#define WG0_OFF  (HS_OFF  + 2u * 16 * SF)      //  98,816
#define WG1_OFF  (WG0_OFF + 2u * 64 * SW2)     // 108,032
#define WU0_OFF  (WG1_OFF + 2u * 64 * SW2)     // 117,248
#define WU1_OFF  (WU0_OFF + 2u * 64 * SW2)     // 126,464
#define TK_OFF   (WU1_OFF + 2u * 64 * SW2)     // 135,680
#define RW_OFF   (TK_OFF + 64u)                // 135,744
#define LDS_BYTES (RW_OFF + 64u)               // 135,808

// ---------------------------------------------------------------------------
// Kernel 3: fused grouped GEMM1 + SwiGLU + GEMM2, atomic accumulate to out.
// Block = 128 threads (4 wave32s), 16 routed tokens of one expert per block.
// Async weight staging is ping-pong double buffered: tile i+1 is issued
// before computing on tile i, so ASYNC latency hides under the WMMAs.
// ---------------------------------------------------------------------------
__launch_bounds__(128)
__global__ void moe_ffn_kernel(float* __restrict__ out) {
  extern __shared__ char smem[];
  __bf16* xs    = (__bf16*)(smem + XS_OFF);
  __bf16* hs    = (__bf16*)(smem + HS_OFF);
  __bf16* wg0   = (__bf16*)(smem + WG0_OFF);
  __bf16* wg1   = (__bf16*)(smem + WG1_OFF);
  __bf16* wu0   = (__bf16*)(smem + WU0_OFF);
  __bf16* wu1   = (__bf16*)(smem + WU1_OFF);
  int*    s_tok = (int*)  (smem + TK_OFF);
  float*  s_rw  = (float*)(smem + RW_OFF);
  const unsigned lbase = __builtin_amdgcn_groupstaticsize();  // dyn-LDS base

  const int e    = blockIdx.x >> 8;       // TT/16 == 256 tiles per expert
  const int tile = blockIdx.x & 255;
  const int cnt  = g_count[e];
  const int m0   = tile * 16;
  if (m0 >= cnt) return;                  // uniform across block
  const int nrows = (cnt - m0 < 16) ? (cnt - m0) : 16;

  const int tid  = threadIdx.x;
  const int lane = tid & 31;              // wave32
  const int wv   = tid >> 5;              // wave id -> 16-wide N subtile
  const int fm   = lane & 15;             // A/C row (M) or B/C col (N)
  const int kh   = lane >> 4;             // K-half select per ISA layout
  const int col  = wv * 16 + fm;

  if (tid < 16) {
    if (tid < nrows) { s_tok[tid] = g_tok[e][m0 + tid]; s_rw[tid] = g_wt[e][m0 + tid]; }
    else             { s_tok[tid] = -1;                 s_rw[tid] = 0.0f; }
  }
  __syncthreads();

  // ---- x tile: async bf16 copy, 16 rows x 1024 (8 threads/row, 16B chunks)
  {
    const int row = tid >> 3;
    const int cb  = (tid & 7) * 128;
    const int t   = s_tok[row];
    if (t >= 0) {
      const __bf16* src = gx + (size_t)t * DD + cb;
      const unsigned l  = lbase + XS_OFF + (unsigned)(row * SD + cb) * 2;
#pragma unroll
      for (int q = 0; q < 16; ++q) async_load16(l + q * 16, src + q * 8);
    } else {
      bf16x8 z = {};
      for (int c = 0; c < 128; c += 8) *(bf16x8*)&xs[row * SD + cb + c] = z;
    }
  }
  wait_async0();
  __syncthreads();

  const int jrow  = tid >> 1;              // staged tile row 0..63
  const int kbase = (tid & 1) * 32;        // 32-element K half of the row
  const unsigned lwg0 = lbase + WG0_OFF + (unsigned)(jrow * SW2 + kbase) * 2;
  const unsigned lwg1 = lbase + WG1_OFF + (unsigned)(jrow * SW2 + kbase) * 2;
  const unsigned lwu0 = lbase + WU0_OFF + (unsigned)(jrow * SW2 + kbase) * 2;
  const unsigned lwu1 = lbase + WU1_OFF + (unsigned)(jrow * SW2 + kbase) * 2;

  // ========== GEMM1: gu = x @ w1[e]^T, SwiGLU -> hs (pipelined) =============
  {
    const __bf16* g1row = gw1 + ((size_t)e * 2 * FF + jrow) * DD + kbase;
    // iteration it = nf_idx*16 + dk_idx; tile = 64 rows x 64 K of gate & up
    auto issue_g1 = [&](int it) {
      const __bf16* pg = g1row + ((size_t)(it >> 4) << 16) + ((it & 15) << 6);
      const __bf16* pu = pg + (size_t)FF * DD;
      const unsigned lg = (it & 1) ? lwg1 : lwg0;
      const unsigned lu = (it & 1) ? lwu1 : lwu0;
#pragma unroll
      for (int q = 0; q < 4; ++q) {
        async_load16(lg + q * 16, pg + q * 8);
        async_load16(lu + q * 16, pu + q * 8);
      }
    };

    issue_g1(0);
    wait_async0();
    __syncthreads();

    v8f accg = {}; v8f accu = {};
#pragma unroll 2
    for (int it = 0; it < 512; ++it) {
      if (it + 1 < 512) issue_g1(it + 1);          // prefetch next tile
      const __bf16* wg = (it & 1) ? wg1 : wg0;
      const __bf16* wu = (it & 1) ? wu1 : wu0;
      const int dk = (it & 15) * 64;
#pragma unroll
      for (int s = 0; s < 2; ++s) {                // two 16x16x32 MACs
        const int ks = dk + s * 32;
        v16bf a  = frag2(&xs[fm * SD + ks + kh * 8],
                         &xs[fm * SD + ks + 16 + kh * 8]);
        v16bf bg = frag2(&wg[col * SW2 + s * 32 + kh * 8],
                         &wg[col * SW2 + s * 32 + 16 + kh * 8]);
        v16bf bu = frag2(&wu[col * SW2 + s * 32 + kh * 8],
                         &wu[col * SW2 + s * 32 + 16 + kh * 8]);
        accg = __builtin_amdgcn_wmma_f32_16x16x32_bf16(false, a, false, bg,
                                                       (short)0, accg, false, false);
        accu = __builtin_amdgcn_wmma_f32_16x16x32_bf16(false, a, false, bu,
                                                       (short)0, accu, false, false);
      }
      if ((it & 15) == 15) {                       // end of K: SwiGLU -> hs
        const int nf = (it >> 4) * 64;
        const int mb = kh * 8;                     // C: VGPR r -> M = r (+8 hi)
#pragma unroll
        for (int r = 0; r < 8; ++r) {
          float g   = accg[r];
          float u   = accu[r];
          float sig = __builtin_amdgcn_rcpf(1.0f + __expf(-g));
          hs[(mb + r) * SF + nf + wv * 16 + fm] = (__bf16)(g * sig * u);
        }
        accg = v8f{}; accu = v8f{};
      }
      wait_async0();                               // next tile resident
      __syncthreads();                             // all waves done reading cur
    }
  }

  // ========== GEMM2: out += route * (h @ w2[e]^T), pipelined + atomic =======
  {
    const __bf16* g2row = gw2 + ((size_t)e * DD + jrow) * FF + kbase;
    // iteration it = nd_idx*32 + fk_idx; tile = 64 rows x 64 K of w2
    auto issue_g2 = [&](int it) {
      const __bf16* p2 = g2row + ((size_t)(it >> 5) << 17) + ((it & 31) << 6);
      const unsigned lw = (it & 1) ? lwg1 : lwg0;
#pragma unroll
      for (int q = 0; q < 4; ++q) async_load16(lw + q * 16, p2 + q * 8);
    };

    issue_g2(0);
    wait_async0();
    __syncthreads();

    v8f acc = {};
#pragma unroll 2
    for (int it = 0; it < 512; ++it) {
      if (it + 1 < 512) issue_g2(it + 1);
      const __bf16* wb = (it & 1) ? wg1 : wg0;
      const int fk = (it & 31) * 64;
#pragma unroll
      for (int s = 0; s < 2; ++s) {
        v16bf a = frag2(&hs[fm * SF + fk + s * 32 + kh * 8],
                        &hs[fm * SF + fk + s * 32 + 16 + kh * 8]);
        v16bf b = frag2(&wb[col * SW2 + s * 32 + kh * 8],
                        &wb[col * SW2 + s * 32 + 16 + kh * 8]);
        acc = __builtin_amdgcn_wmma_f32_16x16x32_bf16(false, a, false, b,
                                                      (short)0, acc, false, false);
      }
      if ((it & 31) == 31) {                       // end of K: scaled atomics
        const int nd = (it >> 5) * 64;
        const int mb = kh * 8;
#pragma unroll
        for (int r = 0; r < 8; ++r) {
          const int m = mb + r;
          if (m < nrows) {
            atomicAdd(&out[(size_t)s_tok[m] * DD + nd + wv * 16 + fm],
                      s_rw[m] * acc[r]);
          }
        }
        acc = v8f{};
      }
      wait_async0();
      __syncthreads();
    }
  }
}

// ---------------------------------------------------------------------------
extern "C" void kernel_launch(void* const* d_in, const int* in_sizes, int n_in,
                              void* d_out, int out_size, void* d_ws, size_t ws_size,
                              hipStream_t stream) {
  (void)in_sizes; (void)n_in; (void)d_ws; (void)ws_size; (void)out_size;
  const float* x      = (const float*)d_in[0];
  const float* gating = (const float*)d_in[1];
  const float* w1     = (const float*)d_in[2];
  const float* w2     = (const float*)d_in[3];
  float* out = (float*)d_out;

  const int n = TT * DD;
  zero_kernel<<<(n + 255) / 256, 256, 0, stream>>>(out, n);

  const size_t nchunks = ((size_t)EE * 2 * FF * DD + (size_t)EE * DD * FF +
                          (size_t)TT * DD) / 8;
  cvt_kernel<<<(unsigned)((nchunks + 255) / 256), 256, 0, stream>>>(x, w1, w2);

  route_kernel<<<(TT + 255) / 256, 256, 0, stream>>>(gating);

  moe_ffn_kernel<<<dim3(EE * (TT / 16)), dim3(128), LDS_BYTES, stream>>>(out);
}
